// TransformerEncoder_23673859735588
// MI455X (gfx1250) — matrix operand order repaired and checked by reading, not compiled, not run on hardware
//
#include <hip/hip_runtime.h>
#include <hip/hip_bf16.h>

// ---------------------------------------------------------------------------
// Types for CDNA5 WMMA (wave32, 16x16x32 bf16 -> f32)
// ---------------------------------------------------------------------------
typedef __bf16 bf16;
typedef __attribute__((ext_vector_type(16))) __bf16 v16bf;
typedef __attribute__((ext_vector_type(8)))  __bf16 v8bf;
typedef __attribute__((ext_vector_type(8)))  float  v8f;

static __device__ __forceinline__ v16bf combine16(v8bf lo, v8bf hi) {
    return __builtin_shufflevector(lo, hi, 0,1,2,3,4,5,6,7,8,9,10,11,12,13,14,15);
}

static __device__ __forceinline__ v8f wmma_bf16(v16bf a, v16bf b, v8f c) {
    // (neg_a, A, neg_b, B, c_mod, C, reuse_a, reuse_b)
    return __builtin_amdgcn_wmma_f32_16x16x32_bf16(false, a, false, b, (short)0, c, false, false);
}

// ---------------------------------------------------------------------------
// 16-lane xor-butterfly reductions via v_permlane16_b32 (VALU, no LDS pipe).
// Packed 4-bit selectors per lane: selector[i] = i ^ mask. bound_ctrl=1 is
// safe (all lanes active, selectors in-row) and lets the tie-copy die.
// ---------------------------------------------------------------------------
#if __has_builtin(__builtin_amdgcn_permlane16)
static __device__ __forceinline__ float pl16(float v, unsigned s0, unsigned s1) {
    int x = __builtin_bit_cast(int, v);
    x = __builtin_amdgcn_permlane16(x, x, (int)s0, (int)s1, false, true);
    return __builtin_bit_cast(float, x);
}
static __device__ __forceinline__ float hmax16(float v) {
    v = fmaxf(v, pl16(v, 0x67452301u, 0xEFCDAB89u));   // xor 1
    v = fmaxf(v, pl16(v, 0x54761032u, 0xDCFE98BAu));   // xor 2
    v = fmaxf(v, pl16(v, 0x32107654u, 0xBA98FEDCu));   // xor 4
    v = fmaxf(v, pl16(v, 0xFEDCBA98u, 0x76543210u));   // xor 8
    return v;
}
static __device__ __forceinline__ float hsum16(float v) {
    v += pl16(v, 0x67452301u, 0xEFCDAB89u);
    v += pl16(v, 0x54761032u, 0xDCFE98BAu);
    v += pl16(v, 0x32107654u, 0xBA98FEDCu);
    v += pl16(v, 0xFEDCBA98u, 0x76543210u);
    return v;
}
#else
static __device__ __forceinline__ float hmax16(float v) {
    #pragma unroll
    for (int m = 1; m < 16; m <<= 1) v = fmaxf(v, __shfl_xor(v, m, 32));
    return v;
}
static __device__ __forceinline__ float hsum16(float v) {
    #pragma unroll
    for (int m = 1; m < 16; m <<= 1) v += __shfl_xor(v, m, 32);
    return v;
}
#endif

// ---------------------------------------------------------------------------
// CDNA5 async global->LDS copy (tracked by ASYNCcnt). Each active lane moves
// 16 bytes from its own global address to its own LDS byte offset.
// ---------------------------------------------------------------------------
static __device__ __forceinline__ unsigned lds_off(const void* p) {
    return (unsigned)(uintptr_t)p;
}
static __device__ __forceinline__ void async_load_b128(unsigned loff, const void* gptr) {
    asm volatile("global_load_async_to_lds_b128 %0, %1, off"
                 :: "v"(loff), "v"(gptr) : "memory");
}
#define WAIT_ASYNC0() asm volatile("s_wait_asynccnt 0" ::: "memory")

// ---------------------------------------------------------------------------
// Problem constants
// ---------------------------------------------------------------------------
#define BB   2
#define SS   2048
#define DD   1024
#define FCC  4096
#define NH   16
#define HD   64
#define ROWS (BB * SS)   // 4096 token rows

// ---------------------------------------------------------------------------
// Weight prep: W[K,N] f32 -> WT[N,K] bf16 (tiled 32x32 transpose through LDS).
// ---------------------------------------------------------------------------
__global__ __launch_bounds__(256)
void transpose_to_bf16(const float* __restrict__ W, bf16* __restrict__ WT,
                       int K, int N)
{
    __shared__ bf16 T[32][33];
    const int tx = threadIdx.x & 31;
    const int ty = threadIdx.x >> 5;           // 0..7
    const int k0 = blockIdx.y * 32;
    const int n0 = blockIdx.x * 32;
    #pragma unroll
    for (int i = 0; i < 4; ++i) {
        const int k = ty + i * 8;
        T[k][tx] = (bf16)W[(size_t)(k0 + k) * N + n0 + tx];
    }
    __syncthreads();
    #pragma unroll
    for (int i = 0; i < 4; ++i) {
        const int n = ty + i * 8;
        WT[(size_t)(n0 + n) * K + k0 + tx] = T[tx][n];
    }
}

// ---------------------------------------------------------------------------
// LayerNorm: one block per row (D=1024, 256 threads -> 4 elems/thread).
// ---------------------------------------------------------------------------
__global__ __launch_bounds__(256)
void ln_kernel(const float* __restrict__ x, const float* __restrict__ add,
               const float* __restrict__ g, const float* __restrict__ bb,
               bf16* __restrict__ outn, float* __restrict__ outres)
{
    const int row  = blockIdx.x;
    const int tid  = threadIdx.x;
    const int lane = tid & 31;
    const int wave = tid >> 5;
    const size_t base = (size_t)row * DD;

    float v[4];
    float s = 0.f, s2 = 0.f;
    const bool has_add = (add != nullptr);
    #pragma unroll
    for (int i = 0; i < 4; ++i) {
        const int c = tid + i * 256;
        float t = x[base + c];
        if (has_add) t += add[base + c];
        v[i] = t;
        s += t; s2 += t * t;
    }
    #pragma unroll
    for (int m = 16; m >= 1; m >>= 1) {
        s  += __shfl_xor(s,  m, 32);
        s2 += __shfl_xor(s2, m, 32);
    }
    __shared__ float red[16];
    if (lane == 0) { red[wave] = s; red[8 + wave] = s2; }
    __syncthreads();
    float ts = 0.f, ts2 = 0.f;
    #pragma unroll
    for (int w = 0; w < 8; ++w) { ts += red[w]; ts2 += red[8 + w]; }
    const float mean = ts * (1.0f / DD);
    const float var  = ts2 * (1.0f / DD) - mean * mean;
    const float rstd = rsqrtf(var + 1e-5f);

    #pragma unroll
    for (int i = 0; i < 4; ++i) {
        const int c = tid + i * 256;
        const float nv = (v[i] - mean) * rstd * g[c] + bb[c];
        outn[base + c] = (bf16)nv;
        if (outres) outres[base + c] = v[i];
    }
}

// ---------------------------------------------------------------------------
// WMMA GEMM:  C[M,N] = A_bf16[M,K] @ BT_bf16[N,K]^T  (+bias [+relu | +res])
// 256 threads = 8 waves, workgroup tile 128x128, K-step 32, double-buffered
// LDS filled exclusively by GLOBAL_LOAD_ASYNC_TO_LDS_B128 (ASYNCcnt).
// EPI: 0 = +bias -> bf16 ; 1 = relu(+bias) -> bf16 ; 2 = +bias +res -> f32
// ---------------------------------------------------------------------------
template <int EPI>
__global__ __launch_bounds__(256)
void gemm_bf16(const bf16*  __restrict__ A,
               const bf16*  __restrict__ BT,
               const float* __restrict__ bias,
               const float* __restrict__ res,
               void* __restrict__ out,
               int M, int N, int K)
{
    __shared__ __align__(16) bf16 As[2][128][40];   // [m][k], padded rows (80B)
    __shared__ __align__(16) bf16 Bt[2][128][40];   // [n][k], padded rows

    const int tid    = threadIdx.x;
    const int lane   = tid & 31;
    const int wave   = tid >> 5;
    const int lane16 = lane & 15;
    const int hi     = lane >> 4;
    const int wm     = wave & 3;
    const int wn     = wave >> 2;
    const int m0     = blockIdx.y * 128;
    const int n0     = blockIdx.x * 128;

    const int c8 = (lane & 3) * 8;           // half offset within a 32-half row

    v8f acc[2][4];
    #pragma unroll
    for (int tm = 0; tm < 2; ++tm)
        #pragma unroll
        for (int tn = 0; tn < 4; ++tn)
            #pragma unroll
            for (int i = 0; i < 8; ++i) acc[tm][tn][i] = 0.f;

    auto issue = [&](int kk, int buf) {
        #pragma unroll
        for (int j2 = 0; j2 < 2; ++j2) {
            const int row = (wave * 2 + j2) * 8 + (lane >> 2);   // 0..127
            async_load_b128(lds_off(&As[buf][row][c8]),
                            A + (size_t)(m0 + row) * K + kk + c8);
            async_load_b128(lds_off(&Bt[buf][row][c8]),
                            BT + (size_t)(n0 + row) * K + kk + c8);
        }
    };

    const int nk = K / 32;
    issue(0, 0);

    for (int kt = 0; kt < nk; ++kt) {
        WAIT_ASYNC0();        // this wave's tile-kt LDS writes landed
        __syncthreads();      // everyone's writes landed; prev compute done
        if (kt + 1 < nk) issue((kt + 1) * 32, (kt + 1) & 1);  // prefetch next
        const int buf = kt & 1;

        v16bf af[2], bfr[4];
        #pragma unroll
        for (int tm = 0; tm < 2; ++tm) {
            // A layout: lane m = lane16; two 8-half runs at k = 8*hi, 16+8*hi
            const bf16* ar = &As[buf][wm * 32 + tm * 16 + lane16][0];
            v8bf lo = *(const v8bf*)(ar + 8 * hi);
            v8bf hh = *(const v8bf*)(ar + 16 + 8 * hi);
            af[tm]  = combine16(lo, hh);
        }
        #pragma unroll
        for (int tn = 0; tn < 4; ++tn) {
            // B layout: lane n = lane16; two 8-half runs at k = 8*hi, 16+8*hi
            const bf16* br = &Bt[buf][wn * 64 + tn * 16 + lane16][0];
            v8bf lo = *(const v8bf*)(br + 8 * hi);
            v8bf hh = *(const v8bf*)(br + 16 + 8 * hi);
            bfr[tn] = combine16(lo, hh);
        }
        #pragma unroll
        for (int tm = 0; tm < 2; ++tm)
            #pragma unroll
            for (int tn = 0; tn < 4; ++tn)
                acc[tm][tn] = wmma_bf16(af[tm], bfr[tn], acc[tm][tn]);
    }

    // ---- epilogue (C layout: row = i + 8*hi, col = lane16) ----
    #pragma unroll
    for (int tm = 0; tm < 2; ++tm) {
        #pragma unroll
        for (int tn = 0; tn < 4; ++tn) {
            #pragma unroll
            for (int i = 0; i < 8; ++i) {
                const int row = m0 + wm * 32 + tm * 16 + i + 8 * hi;
                const int col = n0 + wn * 64 + tn * 16 + lane16;
                float v = acc[tm][tn][i] + bias[col];
                const size_t o = (size_t)row * N + col;
                if (EPI == 0) {
                    ((bf16*)out)[o] = (bf16)v;
                } else if (EPI == 1) {
                    ((bf16*)out)[o] = (bf16)fmaxf(v, 0.f);
                } else {
                    ((float*)out)[o] = v + res[o];
                }
            }
        }
    }
}

// ---------------------------------------------------------------------------
// Flash attention. Grid (S/64, H, B), block = 128 (4 waves).
// 64-key main-loop chunk: the online-softmax fixed costs (O rescale, m/l
// update, butterflies, barriers) amortize over 2x the keys vs a 32-key chunk.
// Q fragments pre-scaled by 1/sqrt(hd). 16 WMMAs per iteration.
// ---------------------------------------------------------------------------
__global__ __launch_bounds__(128)
void attn_kernel(const bf16* __restrict__ qkv, float* __restrict__ attn_out)
{
    const int tid    = threadIdx.x;
    const int wave   = tid >> 5;
    const int lane   = tid & 31;
    const int lane16 = lane & 15;
    const int hi     = lane >> 4;
    const int h      = blockIdx.y;
    const int b      = blockIdx.z;
    const int q0     = blockIdx.x * 64 + wave * 16;

    __shared__ __align__(16) bf16 Vt[HD][72];        // V chunk transposed [hd][key 0..63]
    __shared__ __align__(16) bf16 Ps[4][16][72];     // per-wave P tile [row][key 0..63]

    const size_t rstride = 3 * DD;
    const bf16* qbase = qkv + (size_t)b * SS * rstride + h * HD;
    const bf16* kbase = qbase + DD;
    const bf16* vbase = qbase + 2 * DD;

    // Q fragments (A layout, M=16, K=32 each), pre-scaled by 1/sqrt(hd)
    v16bf qa[2];
    {
        const bf16* qrow = qbase + (size_t)(q0 + lane16) * rstride;
        #pragma unroll
        for (int kc = 0; kc < 2; ++kc) {
            v8bf lo = *(const v8bf*)(qrow + kc * 32 + 8 * hi);
            v8bf hh = *(const v8bf*)(qrow + kc * 32 + 16 + 8 * hi);
            v16bf q = combine16(lo, hh);
            #pragma unroll
            for (int e = 0; e < 16; ++e) q[e] = (bf16)((float)q[e] * 0.125f);
            qa[kc] = q;
        }
    }

    v8f   o[4];
    float mr[8], lr[8];
    #pragma unroll
    for (int t = 0; t < 4; ++t)
        #pragma unroll
        for (int i = 0; i < 8; ++i) o[t][i] = 0.f;
    #pragma unroll
    for (int i = 0; i < 8; ++i) { mr[i] = -3.0e38f; lr[i] = 0.f; }

    for (int kk = 0; kk < SS; kk += 64) {
        __syncthreads();  // previous iteration's LDS reads done

        // ---- stage V chunk transposed: Vt[hd][key], 2 keys packed per b32 ----
        #pragma unroll
        for (int u = 0; u < 2; ++u) {
            const int unit = tid + u * 128;        // 0..255
            const int kp = unit >> 3;              // 0..31 key pairs
            const int hc = (unit & 7) * 8;         // 8 hd values
            const bf16* v0p = vbase + (size_t)(kk + 2 * kp) * rstride + hc;
            v8bf a = *(const v8bf*)v0p;
            v8bf c = *(const v8bf*)(v0p + rstride);
            #pragma unroll
            for (int e = 0; e < 8; ++e) {
                const unsigned lo  = (unsigned)__builtin_bit_cast(unsigned short, a[e]);
                const unsigned hi2 = (unsigned)__builtin_bit_cast(unsigned short, c[e]);
                *(unsigned*)&Vt[hc + e][2 * kp] = lo | (hi2 << 16);
            }
        }

        // ---- scores: four 16x16 tiles (keys kk+16t .. kk+16t+15) ----
        v8f s[4];
        #pragma unroll
        for (int t = 0; t < 4; ++t) {
            #pragma unroll
            for (int i = 0; i < 8; ++i) s[t][i] = 0.f;
            const bf16* krow = kbase + (size_t)(kk + t * 16 + lane16) * rstride;
            #pragma unroll
            for (int kc = 0; kc < 2; ++kc) {
                // B layout: lane n = key, two 8-hd runs at 8*hi, 16+8*hi
                v8bf lo = *(const v8bf*)(krow + kc * 32 + 8 * hi);
                v8bf hh = *(const v8bf*)(krow + kc * 32 + 16 + 8 * hi);
                s[t] = wmma_bf16(qa[kc], combine16(lo, hh), s[t]);
            }
        }

        // ---- online softmax update (row r = i + 8*hi across 16 lanes) ----
        float mnew[8];
        #pragma unroll
        for (int i = 0; i < 8; ++i) {
            const float mx = hmax16(fmaxf(fmaxf(s[0][i], s[1][i]),
                                          fmaxf(s[2][i], s[3][i])));
            mnew[i] = fmaxf(mr[i], mx);
            const float alpha = __expf(mr[i] - mnew[i]);
            mr[i] = mnew[i];
            lr[i] *= alpha;
            #pragma unroll
            for (int t = 0; t < 4; ++t) o[t][i] *= alpha;
        }
        #pragma unroll
        for (int t = 0; t < 4; ++t) {
            #pragma unroll
            for (int i = 0; i < 8; ++i) {
                const float p = __expf(s[t][i] - mnew[i]);
                s[t][i] = p;
                Ps[wave][i + 8 * hi][t * 16 + lane16] = (bf16)p;
            }
        }
        #pragma unroll
        for (int i = 0; i < 8; ++i)
            lr[i] += hsum16((s[0][i] + s[1][i]) + (s[2][i] + s[3][i]));

        __syncthreads();  // Vt + Ps visible

        // ---- P @ V : A = two P tiles (16x32), B = Vt columns ----
        v16bf pa[2];
        #pragma unroll
        for (int f = 0; f < 2; ++f) {
            const bf16* pr = &Ps[wave][lane16][f * 32];
            v8bf lo = *(const v8bf*)(pr + 8 * hi);
            v8bf hh = *(const v8bf*)(pr + 16 + 8 * hi);
            pa[f] = combine16(lo, hh);
        }
        #pragma unroll
        for (int nt = 0; nt < 4; ++nt) {
            const bf16* vr = &Vt[nt * 16 + lane16][0];
            #pragma unroll
            for (int f = 0; f < 2; ++f) {
                v8bf lo = *(const v8bf*)(vr + f * 32 + 8 * hi);
                v8bf hh = *(const v8bf*)(vr + f * 32 + 16 + 8 * hi);
                o[nt] = wmma_bf16(pa[f], combine16(lo, hh), o[nt]);
            }
        }
    }

    // ---- finalize: multiply by 1/l, write f32 ----
    float il[8];
    #pragma unroll
    for (int i = 0; i < 8; ++i) il[i] = 1.0f / lr[i];
    #pragma unroll
    for (int nt = 0; nt < 4; ++nt) {
        #pragma unroll
        for (int i = 0; i < 8; ++i) {
            const int row = q0 + i + 8 * hi;
            const int col = h * HD + nt * 16 + lane16;
            attn_out[((size_t)b * SS + row) * DD + col] = o[nt][i] * il[i];
        }
    }
}

// ---------------------------------------------------------------------------
// Launch: weight prep + 6-kernel pipeline, all on `stream`.
// ---------------------------------------------------------------------------
extern "C" void kernel_launch(void* const* d_in, const int* in_sizes, int n_in,
                              void* d_out, int out_size, void* d_ws, size_t ws_size,
                              hipStream_t stream)
{
    const float* x     = (const float*)d_in[0];
    const float* ln1_g = (const float*)d_in[1];
    const float* ln1_b = (const float*)d_in[2];
    const float* w_qkv = (const float*)d_in[3];
    const float* b_qkv = (const float*)d_in[4];
    const float* ln2_g = (const float*)d_in[5];
    const float* ln2_b = (const float*)d_in[6];
    const float* w1    = (const float*)d_in[7];
    const float* b1    = (const float*)d_in[8];
    const float* w2    = (const float*)d_in[9];
    const float* b2    = (const float*)d_in[10];
    float* out = (float*)d_out;

    // workspace layout
    char* p = (char*)d_ws;
    bf16*  h1       = (bf16*)p;  p += (size_t)ROWS * DD * sizeof(bf16);       //  8.4 MB
    bf16*  qkv      = (bf16*)p;  p += (size_t)ROWS * 3 * DD * sizeof(bf16);   // 25.2 MB
    float* attn     = (float*)p; p += (size_t)ROWS * DD * sizeof(float);      // 16.8 MB
    float* attn_res = (float*)p; p += (size_t)ROWS * DD * sizeof(float);      // 16.8 MB
    bf16*  h2       = (bf16*)p;  p += (size_t)ROWS * DD * sizeof(bf16);       //  8.4 MB
    bf16*  mact     = (bf16*)p;  p += (size_t)ROWS * FCC * sizeof(bf16);      // 33.6 MB
    bf16*  wqkvT    = (bf16*)p;  p += (size_t)3 * DD * DD * sizeof(bf16);     //  6.3 MB
    bf16*  w1T      = (bf16*)p;  p += (size_t)DD * FCC * sizeof(bf16);        //  8.4 MB
    bf16*  w2T      = (bf16*)p;  p += (size_t)FCC * DD * sizeof(bf16);        //  8.4 MB

    // 0) weight prep: f32 [K,N] -> bf16 [N,K]
    transpose_to_bf16<<<dim3(3 * DD / 32, DD / 32),  256, 0, stream>>>(w_qkv, wqkvT, DD,  3 * DD);
    transpose_to_bf16<<<dim3(FCC / 32,    DD / 32),  256, 0, stream>>>(w1,    w1T,   DD,  FCC);
    transpose_to_bf16<<<dim3(DD / 32,     FCC / 32), 256, 0, stream>>>(w2,    w2T,   FCC, DD);

    // 1) LN1: x -> h1 (bf16)
    ln_kernel<<<ROWS, 256, 0, stream>>>(x, nullptr, ln1_g, ln1_b, h1, nullptr);

    // 2) QKV GEMM: [4096,1024] @ [1024,3072] + b_qkv -> qkv (bf16)
    gemm_bf16<0><<<dim3(3 * DD / 128, ROWS / 128), 256, 0, stream>>>(
        h1, wqkvT, b_qkv, nullptr, (void*)qkv, ROWS, 3 * DD, DD);

    // 3) flash attention -> attn (f32, pre-residual)
    attn_kernel<<<dim3(SS / 64, NH, BB), 128, 0, stream>>>(qkv, attn);

    // 4) LN2 over (attn + x): -> h2 (bf16), attn_res = attn + x (f32)
    ln_kernel<<<ROWS, 256, 0, stream>>>(attn, x, ln2_g, ln2_b, h2, attn_res);

    // 5) MLP1: relu([4096,1024] @ [1024,4096] + b1) -> mact (bf16)
    gemm_bf16<1><<<dim3(FCC / 128, ROWS / 128), 256, 0, stream>>>(
        h2, w1T, b1, nullptr, (void*)mact, ROWS, FCC, DD);

    // 6) MLP2: [4096,4096] @ [4096,1024] + b2 + attn_res -> out (f32)
    gemm_bf16<2><<<dim3(DD / 128, ROWS / 128), 256, 0, stream>>>(
        mact, w2T, b2, attn_res, (void*)out, ROWS, DD, FCC);
}
